// CPAttn_87574383165715
// MI455X (gfx1250) — compile-verified, matrix operand order checked by reading, not run on hardware
//
#include <hip/hip_runtime.h>

// ---------------------------------------------------------------------------
// Problem constants (B,S,DIM,H,D) = (2, 2048, 2048, 16, 128)
// ---------------------------------------------------------------------------
constexpr int Bc    = 2;
constexpr int Sc    = 2048;
constexpr int DIMc  = 2048;
constexpr int Hc    = 16;
constexpr int Dc    = 128;
constexpr int INNER = Hc * Dc;        // 2048
constexpr int Mrows = Bc * Sc;        // 4096 token rows

typedef __attribute__((ext_vector_type(16))) _Float16 v16h;
typedef __attribute__((ext_vector_type(8)))  _Float16 h8;
typedef __attribute__((ext_vector_type(8)))  float    v8f;
typedef __attribute__((ext_vector_type(4)))  float    f4;
typedef __attribute__((ext_vector_type(4)))  unsigned u32x4;
typedef __attribute__((ext_vector_type(8)))  int      i32x8;
typedef __attribute__((ext_vector_type(4)))  int      i32x4;

#if defined(__has_builtin)
#if __has_builtin(__builtin_amdgcn_tensor_load_to_lds)
#define HAVE_TDM 1
#endif
#endif
#ifndef HAVE_TDM
#define HAVE_TDM 0
#endif

// ---------------------------------------------------------------------------
// gfx1250 async global->LDS copy (ASYNCcnt) -- assembler-verified round 2.
// LDS byte address = low 32 bits of the generic pointer (aperture | offset).
// ---------------------------------------------------------------------------
__device__ __forceinline__ unsigned lds_off(const void* p) {
  return (unsigned)(unsigned long long)(uintptr_t)p;
}
__device__ __forceinline__ void cp_async_b128(_Float16* ldsDst,
                                              const _Float16* gSrc) {
  const unsigned lds = lds_off(ldsDst);
  const unsigned long long ga = (unsigned long long)(uintptr_t)gSrc;
  asm volatile("global_load_async_to_lds_b128 %0, %1, off"
               :: "v"(lds), "v"(ga)
               : "memory");
}
__device__ __forceinline__ void cp_async_wait0() {
  asm volatile("s_wait_asynccnt 0" ::: "memory");
}

// ---------------------------------------------------------------------------
// Tensor Data Mover: one 2D tile descriptor (D#) per call, issued per-wave.
// Groups built per ISA 8.3/8.4: g0 = {count|flags, lds_addr, gaddr lo,
// gaddr hi | type=2<<30}; g1 packs data_size=2B, tensor dims (huge, no OOB),
// tile_dim0/1 and dim0 stride. TDM writes the tile contiguously into LDS
// (X fastest), matching a row-major [tileY][tileX] buffer exactly.
// ---------------------------------------------------------------------------
__device__ __forceinline__ void tdm_load_2d(unsigned ldsAddr,
                                            const _Float16* gsrc,
                                            unsigned tileX, unsigned tileY,
                                            unsigned long long strideElems) {
#if HAVE_TDM
  const unsigned long long ga = (unsigned long long)(uintptr_t)gsrc;
  const unsigned TD0 = 0x40000000u, TD1 = 0x40000000u;  // no OOB clipping
  u32x4 g0 = {};
  g0[0] = 1u;                                            // count=1 (valid D#)
  g0[1] = ldsAddr;                                       // lds_addr bytes
  g0[2] = (unsigned)(ga & 0xFFFFFFFFu);                  // global_addr[31:0]
  g0[3] = (unsigned)((ga >> 32) & 0x01FFFFFFu) | (2u << 30);  // [56:32]|type=2
  i32x8 g1 = {};
  g1[0] = (int)(1u << 16);                               // data_size=1 -> 2B
  g1[1] = (int)((TD0 & 0xFFFFu) << 16);                  // tensor_dim0 lo16
  g1[2] = (int)(((TD0 >> 16) & 0xFFFFu) | ((TD1 & 0xFFFFu) << 16));
  g1[3] = (int)(((TD1 >> 16) & 0xFFFFu) | (tileX << 16));// tile_dim0
  g1[4] = (int)(tileY & 0xFFFFu);                        // tile_dim1 (dim2=0)
  g1[5] = (int)(unsigned)(strideElems & 0xFFFFFFFFu);    // dim0 stride lo32
  g1[6] = (int)(unsigned)((strideElems >> 32) & 0xFFFFu);// dim0 stride hi16
  g1[7] = 0;
  i32x4 g2 = {}, g3 = {};
#if __clang_major__ >= 23
  i32x8 g4 = {};
  __builtin_amdgcn_tensor_load_to_lds(g0, g1, g2, g3, g4, 0);
#else
  __builtin_amdgcn_tensor_load_to_lds(g0, g1, g2, g3, 0);
#endif
#endif
}
__device__ __forceinline__ void tdm_wait0() {
#if HAVE_TDM
  __builtin_amdgcn_s_wait_tensorcnt(0);
#endif
}

// ---------------------------------------------------------------------------
// Fragment loader for 16-bit WMMA operands (wave32): lane&15 = row; halves
// j map to K = (j>>3)*16 + (lane>>4)*8 + (j&7): two contiguous 16-byte runs.
// ---------------------------------------------------------------------------
__device__ __forceinline__ v16h ld_frag(const _Float16* base, int row,
                                        int stride, int koff, int lane) {
  const int hh = (lane >> 4) * 8;
  const _Float16* p = base + (size_t)row * stride + koff + hh;
  h8 lo = *(const h8*)(p);
  h8 hi = *(const h8*)(p + 16);
  v16h f;
#pragma unroll
  for (int j = 0; j < 8; ++j) { f[j] = lo[j]; f[j + 8] = hi[j]; }
  return f;
}

// ---------------------------------------------------------------------------
// f32 -> f16 bulk convert (one-time). 8 elements per thread.
// ---------------------------------------------------------------------------
__global__ __launch_bounds__(256)
void cvt_f16(const float* __restrict__ in, _Float16* __restrict__ out) {
  const size_t i = ((size_t)blockIdx.x * 256 + threadIdx.x) * 8;
  f4 a = *(const f4*)(in + i);
  f4 b = *(const f4*)(in + i + 4);
  h8 o;
#pragma unroll
  for (int j = 0; j < 4; ++j) { o[j] = (_Float16)a[j]; o[j + 4] = (_Float16)b[j]; }
  *(h8*)(out + i) = o;
}

// ---------------------------------------------------------------------------
// GEMM: C[m,n] = sum_k A[m,k]*W[n,k] + bias[n]; A,W f16 row-major (MxK, NxK).
// Block 256 (8 waves), tile 128x128, K-step 32; wave tile 32x64 -> 8 WMMA.
// Double-buffered LDS; tiles streamed by the Tensor Data Mover (wave 0 issues
// two D# per step, s_wait_tensorcnt gates the buffer swap), overlapping the
// 16KB/step DMA with the WMMA chain.
// ---------------------------------------------------------------------------
template <typename OutT>
__global__ __launch_bounds__(256)
void gemm_f16(const _Float16* __restrict__ A, const _Float16* __restrict__ W,
              const float* __restrict__ bias, OutT* __restrict__ C,
              int Ntot, int Ktot) {
  __shared__ _Float16 sA[2][128 * 32];
  __shared__ _Float16 sB[2][128 * 32];

  const int tid  = threadIdx.x;
  const int lane = tid & 31;
  const int wid  = tid >> 5;
  const int wy   = wid >> 1;       // 0..3 -> M offset wy*32
  const int wx   = wid & 1;        // 0..1 -> N offset wx*64
  const int m0   = blockIdx.y * 128;
  const int n0   = blockIdx.x * 128;

  auto issueTile = [&](int buf, int kt) {
#if HAVE_TDM
    if (wid == 0) {
      tdm_load_2d(lds_off(&sA[buf][0]), A + (size_t)m0 * Ktot + kt, 32, 128,
                  (unsigned long long)Ktot);
      tdm_load_2d(lds_off(&sB[buf][0]), W + (size_t)n0 * Ktot + kt, 32, 128,
                  (unsigned long long)Ktot);
    }
#else
#pragma unroll
    for (int c = 0; c < 2; ++c) {
      const int idx = tid + c * 256;          // 0..511
      const int row = idx >> 2;
      const int col = (idx & 3) * 8;
      cp_async_b128(&sA[buf][row * 32 + col],
                    A + (size_t)(m0 + row) * Ktot + kt + col);
      cp_async_b128(&sB[buf][row * 32 + col],
                    W + (size_t)(n0 + row) * Ktot + kt + col);
    }
#endif
  };
  auto waitTiles = [&]() {
#if HAVE_TDM
    if (wid == 0) tdm_wait0();
#else
    cp_async_wait0();
#endif
    __syncthreads();
  };

  const v8f vzero = {};
  v8f acc[2][4];
#pragma unroll
  for (int i = 0; i < 2; ++i)
#pragma unroll
    for (int j = 0; j < 4; ++j) acc[i][j] = vzero;

  issueTile(0, 0);
  int ib = 0;
  for (int kt = 0; kt < Ktot; kt += 32, ib ^= 1) {
    waitTiles();                       // tile kt landed; prev reads all done
    if (kt + 32 < Ktot) issueTile(ib ^ 1, kt + 32);   // overlap next DMA

    const v16h a0 = ld_frag(sA[ib], wy * 32 + (lane & 15), 32, 0, lane);
    const v16h a1 = ld_frag(sA[ib], wy * 32 + 16 + (lane & 15), 32, 0, lane);
    v16h bf[4];
#pragma unroll
    for (int j = 0; j < 4; ++j)
      bf[j] = ld_frag(sB[ib], wx * 64 + j * 16 + (lane & 15), 32, 0, lane);

#pragma unroll
    for (int j = 0; j < 4; ++j) {
      acc[0][j] = __builtin_amdgcn_wmma_f32_16x16x32_f16(false, a0, false, bf[j],
                                                         (short)0, acc[0][j], false, false);
      acc[1][j] = __builtin_amdgcn_wmma_f32_16x16x32_f16(false, a1, false, bf[j],
                                                         (short)0, acc[1][j], false, false);
    }
  }

  const int nlane = lane & 15;
  const int mh    = (lane >> 4) * 8;
#pragma unroll
  for (int i = 0; i < 2; ++i)
#pragma unroll
    for (int j = 0; j < 4; ++j) {
      const int col = n0 + wx * 64 + j * 16 + nlane;
      const float bv = bias[col];
#pragma unroll
      for (int r = 0; r < 8; ++r) {
        const int row = m0 + wy * 32 + i * 16 + mh + r;
        C[(size_t)row * Ntot + col] = (OutT)(acc[i][j][r] + bv);
      }
    }
}

// ---------------------------------------------------------------------------
// Kernel 2: per-token RMSNorm (full INNER, as in reference) + RoPE.
//   q16,k16 : [b,h,s,d] ; vT : [b,h,d,s]
// ---------------------------------------------------------------------------
__global__ __launch_bounds__(256)
void norm_rope(const _Float16* __restrict__ qf, const _Float16* __restrict__ kf,
               const _Float16* __restrict__ vf, const float* __restrict__ gq,
               const float* __restrict__ gk, const float* __restrict__ rcl,
               const float* __restrict__ rsl, const float* __restrict__ rcf,
               const float* __restrict__ rsf, _Float16* __restrict__ q16,
               _Float16* __restrict__ k16, _Float16* __restrict__ vT) {
  const int tok = blockIdx.x;
  const int b   = tok / Sc;
  const int s   = tok % Sc;
  const int tid = threadIdx.x;
  const int lane = tid & 31, wid = tid >> 5;
  const int idx0 = tid * 8;
  const int hh   = idx0 >> 7;
  const int dl   = idx0 & 127;

  __shared__ float red[8];
  const size_t rowOff = (size_t)tok * INNER;

  for (int which = 0; which < 2; ++which) {
    const _Float16* src = which == 0 ? qf : kf;
    const float* g   = which == 0 ? gq : gk;
    const float* rc  = (which == 0 ? rcl : rcf) + (size_t)tok * Dc;
    const float* rs  = (which == 0 ? rsl : rsf) + (size_t)tok * Dc;
    _Float16*    dst = which == 0 ? q16 : k16;

    float vals[8];
    float ss = 0.f;
#pragma unroll
    for (int j = 0; j < 8; ++j) {
      vals[j] = (float)src[rowOff + idx0 + j];
      ss += vals[j] * vals[j];
    }
#pragma unroll
    for (int off = 16; off >= 1; off >>= 1) ss += __shfl_xor(ss, off, 32);
    if (lane == 0) red[wid] = ss;
    __syncthreads();
    float tot = 0.f;
#pragma unroll
    for (int i = 0; i < 8; ++i) tot += red[i];
    __syncthreads();
    const float rms = rsqrtf(tot * (1.0f / INNER) + 1e-6f);

    _Float16 out[8];
#pragma unroll
    for (int p = 0; p < 4; ++p) {
      const int e = 2 * p;
      const float xe = vals[e]     * rms * g[idx0 + e];
      const float xo = vals[e + 1] * rms * g[idx0 + e + 1];
      const float ce = rc[dl + e];
      const float so = rs[dl + e + 1];
      out[e]     = (_Float16)(xe * ce - xo * so);
      out[e + 1] = (_Float16)(xe * so + xo * ce);
    }
    const size_t base = ((size_t)(b * Hc + hh) * Sc + s) * Dc + dl;
#pragma unroll
    for (int j = 0; j < 8; ++j) dst[base + j] = out[j];
  }

#pragma unroll
  for (int j = 0; j < 8; ++j) {
    vT[((size_t)(b * Hc + hh) * Dc + (dl + j)) * Sc + s] = vf[rowOff + idx0 + j];
  }
}

// ---------------------------------------------------------------------------
// Kernel 3: flash attention per (b,h). 8 waves x 16 query rows per block.
//   S^T = K.Q^T ; O^T += V^T.P^T  (C-layout of S^T == B-layout of P^T:
// softmax->PV hand-off is purely in-lane). K/V tiles double-buffered through
// LDS with async global->LDS copies overlapping the WMMA chain.
// ---------------------------------------------------------------------------
__global__ __launch_bounds__(256)
void attn_fa(const _Float16* __restrict__ q16, const _Float16* __restrict__ k16,
             const _Float16* __restrict__ vT, _Float16* __restrict__ attn16) {
  __shared__ _Float16 sK[2][32 * 128];    // [key][d]
  __shared__ _Float16 sVT[2][128 * 32];   // [d][key]

  const int tid  = threadIdx.x;
  const int lane = tid & 31;
  const int wid  = tid >> 5;
  const int bh   = blockIdx.y;
  const int b    = bh >> 4;
  const int h    = bh & 15;
  const int qrow = blockIdx.x * 128 + wid * 16 + (lane & 15);
  const float scale = 0.08838834764831845f;    // 1/sqrt(128)

  const _Float16* qp = q16 + ((size_t)bh * Sc + qrow) * Dc;
  v16h bq[4];
#pragma unroll
  for (int f = 0; f < 4; ++f) bq[f] = ld_frag(qp, 0, 0, f * 32, lane);

  auto issueKV = [&](int buf, int kb) {
#pragma unroll
    for (int c0 = 0; c0 < 2; ++c0) {
      const int c = tid + c0 * 256;
      const int key = c >> 4, dd = (c & 15) * 8;
      cp_async_b128(&sK[buf][key * 128 + dd],
                    k16 + ((size_t)bh * Sc + kb + key) * Dc + dd);
      const int d = c >> 2, kk = (c & 3) * 8;
      cp_async_b128(&sVT[buf][d * 32 + kk],
                    vT + ((size_t)bh * Dc + d) * Sc + kb + kk);
    }
  };

  const v8f vzero = {};
  v8f oacc[8];
#pragma unroll
  for (int dt = 0; dt < 8; ++dt) oacc[dt] = vzero;
  float m = -3.0e38f;
  float l = 0.f;

  issueKV(0, 0);
  int ib = 0;
  for (int kb = 0; kb < Sc; kb += 32, ib ^= 1) {
    cp_async_wait0();
    __syncthreads();
    if (kb + 32 < Sc) issueKV(ib ^ 1, kb + 32);   // overlap next copies

    v8f sc[2];
#pragma unroll
    for (int mt = 0; mt < 2; ++mt) {
      sc[mt] = vzero;
#pragma unroll
      for (int f = 0; f < 4; ++f) {
        const v16h aK = ld_frag(sK[ib], mt * 16 + (lane & 15), 128, f * 32, lane);
        sc[mt] = __builtin_amdgcn_wmma_f32_16x16x32_f16(false, aK, false, bq[f],
                                                        (short)0, sc[mt], false, false);
      }
    }

    float mloc = -3.0e38f;
#pragma unroll
    for (int mt = 0; mt < 2; ++mt)
#pragma unroll
      for (int r = 0; r < 8; ++r) {
        const float sv = sc[mt][r] * scale;
        sc[mt][r] = sv;
        mloc = fmaxf(mloc, sv);
      }
    mloc = fmaxf(mloc, __shfl_xor(mloc, 16, 32));
    const float mnew  = fmaxf(m, mloc);
    const float alpha = __expf(m - mnew);

    float rowsum = 0.f;
    v16h bP;
#pragma unroll
    for (int mt = 0; mt < 2; ++mt)
#pragma unroll
      for (int r = 0; r < 8; ++r) {
        const float p = __expf(sc[mt][r] - mnew);
        rowsum += p;
        bP[mt * 8 + r] = (_Float16)p;   // in-lane C->B repack
      }
    rowsum += __shfl_xor(rowsum, 16, 32);
    l = l * alpha + rowsum;
    m = mnew;

#pragma unroll
    for (int dt = 0; dt < 8; ++dt) {
#pragma unroll
      for (int r = 0; r < 8; ++r) oacc[dt][r] *= alpha;
      const v16h aV = ld_frag(sVT[ib], dt * 16 + (lane & 15), 32, 0, lane);
      oacc[dt] = __builtin_amdgcn_wmma_f32_16x16x32_f16(false, aV, false, bP,
                                                        (short)0, oacc[dt], false, false);
    }
  }

  const float inv = 1.0f / l;
  const int mh = (lane >> 4) * 8;
#pragma unroll
  for (int dt = 0; dt < 8; ++dt)
#pragma unroll
    for (int r = 0; r < 8; ++r) {
      const int d = dt * 16 + mh + r;
      attn16[((size_t)b * Sc + qrow) * INNER + h * Dc + d] =
          (_Float16)(oacc[dt][r] * inv);
    }
}

// ---------------------------------------------------------------------------
// Host launcher. Inputs: 0:x 1:rcl 2:rsl 3:rcf 4:rsf 5:wq 6:bq 7:wk 8:bk
//                        9:wv 10:bv 11:wo 12:bo 13:gq 14:gk
// ---------------------------------------------------------------------------
extern "C" void kernel_launch(void* const* d_in, const int* in_sizes, int n_in,
                              void* d_out, int out_size, void* d_ws, size_t ws_size,
                              hipStream_t stream) {
  const float* x   = (const float*)d_in[0];
  const float* rcl = (const float*)d_in[1];
  const float* rsl = (const float*)d_in[2];
  const float* rcf = (const float*)d_in[3];
  const float* rsf = (const float*)d_in[4];
  const float* wq  = (const float*)d_in[5];
  const float* bq  = (const float*)d_in[6];
  const float* wk  = (const float*)d_in[7];
  const float* bk  = (const float*)d_in[8];
  const float* wv  = (const float*)d_in[9];
  const float* bv  = (const float*)d_in[10];
  const float* wo  = (const float*)d_in[11];
  const float* bo  = (const float*)d_in[12];
  const float* gq  = (const float*)d_in[13];
  const float* gk  = (const float*)d_in[14];

  char* wsb = (char*)d_ws;
  const size_t tokmat = (size_t)Mrows * INNER * sizeof(_Float16);  // 16.8 MB
  const size_t wmat   = (size_t)INNER * DIMc * sizeof(_Float16);   //  8.4 MB

  _Float16* x16  = (_Float16*)(wsb);
  _Float16* w16q = (_Float16*)(wsb + tokmat);
  _Float16* w16k = (_Float16*)(wsb + tokmat + wmat);
  _Float16* w16v = (_Float16*)(wsb + tokmat + 2 * wmat);
  _Float16* w16o = (_Float16*)(wsb + tokmat + 3 * wmat);
  char*     base1 = wsb + tokmat + 4 * wmat;
  _Float16* qh   = (_Float16*)(base1);
  _Float16* kh   = (_Float16*)(base1 + tokmat);
  _Float16* vh   = (_Float16*)(base1 + 2 * tokmat);
  _Float16* q16  = (_Float16*)(base1 + 3 * tokmat);
  _Float16* k16  = (_Float16*)(base1 + 4 * tokmat);
  _Float16* vT   = (_Float16*)(base1 + 5 * tokmat);
  _Float16* att  = qh;   // alias dead qh region after norm_rope

  cvt_f16<<<(Mrows * INNER) / (256 * 8), 256, 0, stream>>>(x, x16);
  cvt_f16<<<(INNER * DIMc) / (256 * 8), 256, 0, stream>>>(wq, w16q);
  cvt_f16<<<(INNER * DIMc) / (256 * 8), 256, 0, stream>>>(wk, w16k);
  cvt_f16<<<(INNER * DIMc) / (256 * 8), 256, 0, stream>>>(wv, w16v);
  cvt_f16<<<(INNER * DIMc) / (256 * 8), 256, 0, stream>>>(wo, w16o);

  const dim3 gg(INNER / 128, Mrows / 128);   // (16, 32)

  gemm_f16<_Float16><<<gg, 256, 0, stream>>>(x16, w16q, bq, qh, INNER, DIMc);
  gemm_f16<_Float16><<<gg, 256, 0, stream>>>(x16, w16k, bk, kh, INNER, DIMc);
  gemm_f16<_Float16><<<gg, 256, 0, stream>>>(x16, w16v, bv, vh, INNER, DIMc);

  norm_rope<<<Mrows, 256, 0, stream>>>(qh, kh, vh, gq, gk, rcl, rsl, rcf, rsf,
                                       q16, k16, vT);

  attn_fa<<<dim3(Sc / 128, Bc * Hc), 256, 0, stream>>>(q16, k16, vT, att);

  gemm_f16<float><<<gg, 256, 0, stream>>>(att, w16o, bo, (float*)d_out,
                                          DIMc, INNER);
}
